// KernelPropagation_24206435681031
// MI455X (gfx1250) — matrix operand order repaired and checked by reading, not compiled, not run on hardware
//
#include <hip/hip_runtime.h>
#include <hip/hip_bf16.h>

typedef __attribute__((ext_vector_type(16))) _Float16 v16h;
typedef __attribute__((ext_vector_type(8)))  float    v8f;

#define M_PTS 1024
#define NB    2
#define NC    64
#define KS    13
#define NA    20
#define PAIRS (KS * NA)      // 260
#define DIMO  64
#define NCNA  (NC * NA)      // 1280
#define R2    0.16f          // RADIUS^2
#define NEG_INV_2SIG (-6.25f) // -1/(2*0.08)

// ---------------------------------------------------------------------------
// Kernel 1: per-(batch, center) gaussian accumulation.
// One block = one (bb, n). 256 threads cover the 260 (ks,na) kernel positions.
// Fixed i-order summation => deterministic FP results across replays.
// ---------------------------------------------------------------------------
__global__ __launch_bounds__(256) void wts_accum_kernel(
    const float* __restrict__ frag,     // (1024, 3)
    const float* __restrict__ clouds,   // (2, 3, 64)
    const float* __restrict__ kernels,  // (13, 20, 3)
    float* __restrict__ wts)            // (2, 13, 1280) scratch
{
    __shared__ float sx[M_PTS], sy[M_PTS], sz[M_PTS];
    __shared__ unsigned smask[M_PTS / 32];
    __shared__ int scnt;

    const int t  = threadIdx.x;
    const int bb = blockIdx.x / NC;
    const int n  = blockIdx.x % NC;

    // Prefetch the kernel-point table (used only in phase B, after staging):
    // 260*3 floats = 3120 B, one cacheline-ish chunk per low thread.
    if (t < 13) __builtin_prefetch(kernels + t * 64, 0, 0);

    if (t == 0) scnt = 0;
    if (t < M_PTS / 32) smask[t] = 0u;
    __syncthreads();

    const float cx = clouds[bb * 3 * NC + 0 * NC + n];
    const float cy = clouds[bb * 3 * NC + 1 * NC + n];
    const float cz = clouds[bb * 3 * NC + 2 * NC + n];

    int local = 0;
    for (int i = t; i < M_PTS; i += 256) {
        float px = frag[i * 3 + 0];
        float py = frag[i * 3 + 1];
        float pz = frag[i * 3 + 2];
        sx[i] = px; sy[i] = py; sz[i] = pz;
        float dx = px - cx, dy = py - cy, dz = pz - cz;
        float d2 = dx * dx + dy * dy + dz * dz;
        if (d2 < R2) { atomicOr(&smask[i >> 5], 1u << (i & 31)); ++local; }
    }
    atomicAdd(&scnt, local);
    __syncthreads();

    // Each thread owns pair j0 = t, and (first 4 threads) j1 = t + 256.
    const int  j0   = t;
    const int  j1   = t + 256;
    const bool has1 = (j1 < PAIRS);

    const float k0x = cx + kernels[j0 * 3 + 0];
    const float k0y = cy + kernels[j0 * 3 + 1];
    const float k0z = cz + kernels[j0 * 3 + 2];
    const int   j1c = has1 ? j1 : 0;
    const float k1x = cx + kernels[j1c * 3 + 0];
    const float k1y = cy + kernels[j1c * 3 + 1];
    const float k1z = cz + kernels[j1c * 3 + 2];

    float acc0 = 0.f, acc1 = 0.f;
    for (int i = 0; i < M_PTS; ++i) {
        // Uniform across the block: cheap branch skips ~73% of points.
        if (!((smask[i >> 5] >> (i & 31)) & 1u)) continue;
        float px = sx[i], py = sy[i], pz = sz[i];
        float dx = px - k0x, dy = py - k0y, dz = pz - k0z;
        acc0 += __expf((dx * dx + dy * dy + dz * dz) * NEG_INV_2SIG);
        if (has1) {
            float ex = px - k1x, ey = py - k1y, ez = pz - k1z;
            acc1 += __expf((ex * ex + ey * ey + ez * ez) * NEG_INV_2SIG);
        }
    }

    const float norm = 1.0f / ((float)scnt + 1.0f);
    // wts layout: [bb][k][n*NA + a], with j = k*NA + a
    {
        int k = j0 / NA, a = j0 % NA;
        wts[(bb * KS + k) * NCNA + n * NA + a] = acc0 * norm;
    }
    if (has1) {
        int k = j1 / NA, a = j1 % NA;
        wts[(bb * KS + k) * NCNA + n * NA + a] = acc1 * norm;
    }
}

// ---------------------------------------------------------------------------
// Kernel 2: feats = W(64x13) @ wts(b,13,1280) via V_WMMA_F32_16X16X32_F16.
// 8 waves per block, one 16x16 output tile per wave; K padded 13 -> 32.
// All loads are unconditional (clamped index) + v_cndmask select: no EXEC
// branching, so EXEC is all-1s at the WMMA as the ISA requires.
// ---------------------------------------------------------------------------
__global__ __launch_bounds__(256) void conv_wmma_kernel(
    const float* __restrict__ W,    // (64, 13)
    const float* __restrict__ wts,  // (2, 13, 1280)
    float* __restrict__ out)        // (2, 64, 1280)
{
    const int lane = threadIdx.x & 31;
    const int tile = blockIdx.x * 8 + (threadIdx.x >> 5);  // 0..639
    const int bb   = tile / 320;
    const int rem  = tile % 320;
    const int mt   = rem / 80;          // 0..3   (M tile)
    const int nt   = rem % 80;          // 0..79  (N tile)

    const int half = lane >> 4;         // 0 or 1
    const int l16  = lane & 15;

    // A matrix (16x32 f16): lane<16 -> M=l16, K in {0..7, 16..23};
    //                       lane>=16 -> M=l16, K in {8..15, 24..31}.
    v16h a;
    const int Mrow = mt * 16 + l16;
    #pragma unroll
    for (int e = 0; e < 16; ++e) {
        const int K  = (e & 7) + (half << 3) + ((e >> 3) << 4);
        const int Kc = (K < KS) ? K : 0;                 // always in-bounds
        float v = W[Mrow * KS + Kc];
        a[e] = (_Float16)((K < KS) ? v : 0.0f);          // v_cndmask, no branch
    }

    // B matrix (32x16 f16): lanes 0-15 hold K in {0..7, 16..23} of column N=l16,
    //                       lanes 16-31 hold K in {8..15, 24..31}.
    v16h bmat;
    const int Ncol = nt * 16 + l16;
    #pragma unroll
    for (int e = 0; e < 16; ++e) {
        const int K  = (e & 7) + (half << 3) + ((e >> 3) << 4);
        const int Kc = (K < KS) ? K : 0;
        float v = wts[(bb * KS + Kc) * NCNA + Ncol];
        bmat[e] = (_Float16)((K < KS) ? v : 0.0f);
    }

    v8f c = {};
    c = __builtin_amdgcn_wmma_f32_16x16x32_f16(
        /*neg_a=*/false, a, /*neg_b=*/false, bmat,
        /*c_mod=*/(short)0, c, /*reuse_a=*/false, /*reuse_b=*/false);

    // D layout: VGPR r holds M = r + 8*half, N = l16.
    #pragma unroll
    for (int r = 0; r < 8; ++r) {
        const int Mo = mt * 16 + (half << 3) + r;
        out[(bb * DIMO + Mo) * NCNA + Ncol] = c[r];
    }
}

// ---------------------------------------------------------------------------
extern "C" void kernel_launch(void* const* d_in, const int* in_sizes, int n_in,
                              void* d_out, int out_size, void* d_ws, size_t ws_size,
                              hipStream_t stream) {
    (void)in_sizes; (void)n_in; (void)out_size; (void)ws_size;
    const float* frag    = (const float*)d_in[0];  // (1024,3)
    const float* clouds  = (const float*)d_in[1];  // (2,3,64)
    const float* kernels = (const float*)d_in[2];  // (13,20,3)
    const float* W       = (const float*)d_in[3];  // (64,13)
    float* out = (float*)d_out;                    // (2,64,64,20)
    float* wts = (float*)d_ws;                     // needs 2*13*1280*4 = 133120 B

    wts_accum_kernel<<<NB * NC, 256, 0, stream>>>(frag, clouds, kernels, wts);
    conv_wmma_kernel<<<(NB * 4 * 80) / 8, 256, 0, stream>>>(W, wts, out);
}
